// Self_Attention_13159779795154
// MI455X (gfx1250) — matrix operand order repaired and checked by reading, not compiled, not run on hardware
//
#include <hip/hip_runtime.h>
#include <hip/hip_bf16.h>

// ---------------------------------------------------------------------------
// CDNA5 (gfx1250) spatial-reduction self-attention, bf16 WMMA everywhere.
// B=4, N=4096, C=512, heads=8, hd=64, 2x2 reduction -> L=1024 keys.
// Pipelining via single-buffered operands: reloads are issued right AFTER the
// WMMAs that consume them (WAR pins the order), so loads for step k+1 overlap
// step k's WMMAs with no extra register pressure and no spills.
// ---------------------------------------------------------------------------

typedef __bf16 bf16_t;
typedef __attribute__((ext_vector_type(16))) __bf16 v16bf;
typedef __attribute__((ext_vector_type(8)))  __bf16 bf16x8;
typedef __attribute__((ext_vector_type(8)))  float   v8f;
typedef __attribute__((ext_vector_type(4)))  float   f32x4;

union ABReg { v16bf v; bf16x8 h[2]; };

__device__ __forceinline__ bf16x8 ld8(const bf16_t* p) { return *(const bf16x8*)p; }

#define WMMA_BF16(A, B, C) \
  __builtin_amdgcn_wmma_f32_16x16x32_bf16(false, (A), false, (B), (short)0, (C), false, false)

// ---------------------------------------------------------------------------
// vectorized f32 -> bf16 (8 elements / thread, b128 in / b128 out)
__global__ void k_cvt8(const float* __restrict__ s, bf16_t* __restrict__ d, int n8) {
  int i = blockIdx.x * blockDim.x + threadIdx.x;
  if (i >= n8) return;
  f32x4 lo = *(const f32x4*)(s + (size_t)i * 8);
  f32x4 hi = *(const f32x4*)(s + (size_t)i * 8 + 4);
  bf16x8 o;
#pragma unroll
  for (int t = 0; t < 4; ++t) { o[t] = (bf16_t)lo[t]; o[t + 4] = (bf16_t)hi[t]; }
  *(bf16x8*)(d + (size_t)i * 8) = o;
}

// repack conv weight [d, c, i, j] (f32) -> [d, ij, c] (bf16); 8 outputs/thread
__global__ void k_packw8(const float* __restrict__ s, bf16_t* __restrict__ d) {
  int o = (blockIdx.x * blockDim.x + threadIdx.x) * 8;   // 512*2048 total, exact
  int dd = o >> 11, rem = o & 2047, ij = rem >> 9, c0 = rem & 511;
  const float* sp = s + dd * 2048 + ij;
  bf16x8 v;
#pragma unroll
  for (int t = 0; t < 8; ++t) v[t] = (bf16_t)sp[(c0 + t) * 4];
  *(bf16x8*)(d + o) = v;
}

// ---------------------------------------------------------------------------
// WMMA GEMM: each wave computes a 16(M) x 64(N) tile, K in chunks of 32.
// A: row-major [M][K] bf16 (MODE 1/2: gathered rows of k/v pre-activations).
// Bt: row-major [N][K] bf16 (B^T; the reference weights are already this shape).
// MODE: 0 = QKV epilogue (scatter Q*scale / Kpre / Vpre)
//       1 = K-reduce: LayerNorm then write K   [b,h,L,64]
//       2 = V-reduce: LayerNorm then write V^T [b,h,64,L]
//       3 = proj: +bias, f32 output
template <int MODE>
__global__ void k_gemm(const bf16_t* __restrict__ A, const bf16_t* __restrict__ Bt,
                       int Kdim, int Ntiles4,
                       bf16_t* __restrict__ oq, bf16_t* __restrict__ ok,
                       bf16_t* __restrict__ ov,
                       const float* __restrict__ gamma, const float* __restrict__ beta,
                       bf16_t* __restrict__ oln,
                       const float* __restrict__ bias, float* __restrict__ of)
{
  const int lane = threadIdx.x & 31;
  const int g    = lane >> 4;                       // half-wave select
  const int w    = blockIdx.x * (blockDim.x >> 5) + (threadIdx.x >> 5);
  const int nt   = w % Ntiles4;                     // 64-wide N tile (== head, modes 1/2)
  const int m0   = (w / Ntiles4) * 16;
  const int n0   = nt * 64;
  const int arow = m0 + (lane & 15);
  const bf16_t* brow = Bt + (size_t)(n0 + (lane & 15)) * Kdim + g * 16;

  auto loadA = [&](int kc, ABReg& a) {
    const bf16_t* p;
    if (MODE == 1 || MODE == 2) {
      // gather: GEMM-K index = ij*512 + c ; row m = b*1024 + p ; p = pr*32+pc
      int b = arow >> 10, pp = arow & 1023;
      int ij = kc >> 4, c0 = (kc & 15) << 5;
      int ii = ij >> 1, jj = ij & 1;
      int n = ((pp >> 5) * 2 + ii) * 64 + (pp & 31) * 2 + jj;
      p = A + ((size_t)(b * 4096 + n) * 512 + c0);
    } else {
      p = A + (size_t)arow * Kdim + kc * 32;
    }
    a.h[0] = ld8(p + g * 16);
    a.h[1] = ld8(p + g * 16 + 8);
  };
  auto loadB = [&](int kc, ABReg (&b)[4]) {
    const bf16_t* p = brow + kc * 32;
#pragma unroll
    for (int j = 0; j < 4; ++j) {
      b[j].h[0] = ld8(p + (size_t)16 * j * Kdim);
      b[j].h[1] = ld8(p + (size_t)16 * j * Kdim + 8);
    }
  };

  v8f acc[4] = {v8f{}, v8f{}, v8f{}, v8f{}};
  const int nk = Kdim >> 5;

  ABReg a, b[4];
  loadA(0, a);
  loadB(0, b);
  for (int kc = 0; kc < nk - 1; ++kc) {
#pragma unroll
    for (int j = 0; j < 4; ++j) acc[j] = WMMA_BF16(a.v, b[j].v, acc[j]);
    loadA(kc + 1, a);                               // WAR: must issue after WMMAs,
    loadB(kc + 1, b);                               // flies while they execute
  }
#pragma unroll
  for (int j = 0; j < 4; ++j) acc[j] = WMMA_BF16(a.v, b[j].v, acc[j]);

  // ------------------------------- epilogues -------------------------------
  if constexpr (MODE == 0) {
    const int row0 = m0 + 8 * g;
    if (n0 < 512) {                                 // Q segment, fold in hd^-0.5
#pragma unroll
      for (int j = 0; j < 4; ++j) {
        int d = n0 + 16 * j + (lane & 15);
        int h = d >> 6, hd = d & 63;
#pragma unroll
        for (int r = 0; r < 8; ++r) {
          int row = row0 + r;                       // b*4096 + n
          int bb = row >> 12, nn = row & 4095;
          oq[((size_t)(bb * 8 + h) * 4096 + nn) * 64 + hd] = (bf16_t)(acc[j][r] * 0.125f);
        }
      }
    } else if (n0 < 1024) {
#pragma unroll
      for (int j = 0; j < 4; ++j) {
        int d = n0 - 512 + 16 * j + (lane & 15);
#pragma unroll
        for (int r = 0; r < 8; ++r)
          ok[(size_t)(row0 + r) * 512 + d] = (bf16_t)acc[j][r];
      }
    } else {
#pragma unroll
      for (int j = 0; j < 4; ++j) {
        int d = n0 - 1024 + 16 * j + (lane & 15);
#pragma unroll
        for (int r = 0; r < 8; ++r)
          ov[(size_t)(row0 + r) * 512 + d] = (bf16_t)acc[j][r];
      }
    }
  } else if constexpr (MODE == 1 || MODE == 2) {
    // LayerNorm over the 64 columns of this tile (== one head), per row.
#pragma unroll
    for (int r = 0; r < 8; ++r) {
      float s = 0.f, ss = 0.f;
#pragma unroll
      for (int j = 0; j < 4; ++j) { float v = acc[j][r]; s += v; ss += v * v; }
      for (int msk = 1; msk < 16; msk <<= 1) {      // reduce within 16-lane half
        s  += __shfl_xor(s,  msk);
        ss += __shfl_xor(ss, msk);
      }
      float mean = s * (1.f / 64.f);
      float var  = ss * (1.f / 64.f) - mean * mean;
      float rstd = rsqrtf(var + 1e-5f);
      int row = m0 + r + 8 * g;                     // b*1024 + p
      int bb = row >> 10, p = row & 1023;
#pragma unroll
      for (int j = 0; j < 4; ++j) {
        int hd = 16 * j + (lane & 15);
        float v = (acc[j][r] - mean) * rstd * gamma[hd] + beta[hd];
        if constexpr (MODE == 1)  // K  [b,h,L,64]
          oln[((size_t)(bb * 8 + nt) * 1024 + p) * 64 + hd] = (bf16_t)v;
        else                      // V^T [b,h,64,L]
          oln[((size_t)(bb * 8 + nt) * 64 + hd) * 1024 + p] = (bf16_t)v;
      }
    }
  } else {
#pragma unroll
    for (int j = 0; j < 4; ++j) {
      int d = n0 + 16 * j + (lane & 15);
      float bz = bias[d];
#pragma unroll
      for (int r = 0; r < 8; ++r) {
        int row = m0 + r + 8 * g;
        of[(size_t)row * 512 + d] = acc[j][r] + bz;
      }
    }
  }
}

// ---------------------------------------------------------------------------
// Flash attention. One wave owns 16 queries of one (b,h); iterates 32-key
// blocks. S^T = K·Q^T (softmax stats are per-lane), O^T = V^T·P^T.
// K operands are single-buffered: reloaded for block kb+1 right after the
// S-tile WMMAs consume them, overlapping softmax + P·V with the loads.
__global__ void k_attn(const bf16_t* __restrict__ qbf, const bf16_t* __restrict__ kbf,
                       const bf16_t* __restrict__ vt, bf16_t* __restrict__ obf)
{
  __shared__ __align__(16) bf16_t pbuf[8][16][48];  // per-wave P staging [q][key]
  const int lane = threadIdx.x & 31, g = lane >> 4, widx = threadIdx.x >> 5;
  const int w = blockIdx.x * 8 + widx;
  const int qblk = w & 255, bh = w >> 8;

  const bf16_t* Q  = qbf + ((size_t)bh * 4096 + qblk * 16) * 64;
  const bf16_t* Kp = kbf + (size_t)bh * 1024 * 64;
  const bf16_t* Vp = vt  + (size_t)bh * 64 * 1024;

  // Q as B-operand (B^T = Q row-major), two hd-chunks, reused for all blocks
  ABReg bq[2];
#pragma unroll
  for (int kc = 0; kc < 2; ++kc) {
    const bf16_t* p = Q + (lane & 15) * 64 + kc * 32 + g * 8;
    bq[kc].h[0] = ld8(p);
    bq[kc].h[1] = ld8(p + 16);
  }

  auto loadK = [&](int kb, ABReg (&ka)[2][2]) {
#pragma unroll
    for (int t = 0; t < 2; ++t)
#pragma unroll
      for (int kc = 0; kc < 2; ++kc) {
        const bf16_t* p = Kp + (size_t)(kb * 32 + t * 16 + (lane & 15)) * 64 + kc * 32 + g * 8;
        ka[t][kc].h[0] = ld8(p);
        ka[t][kc].h[1] = ld8(p + 16);
      }
  };
  auto loadV = [&](int kb, int j, ABReg& a) {
    const bf16_t* p = Vp + (size_t)(16 * j + (lane & 15)) * 1024 + kb * 32 + g * 8;
    a.h[0] = ld8(p);
    a.h[1] = ld8(p + 16);
  };

  v8f ot[4] = {v8f{}, v8f{}, v8f{}, v8f{}};          // O^T, 4 hd tiles
  float mrun = -3.0e38f, lrun = 0.f;

  ABReg ka[2][2];
  loadK(0, ka);
  for (int kb = 0; kb < 32; ++kb) {
    // ---- S^T: two 16-key tiles
    v8f st[2];
#pragma unroll
    for (int t = 0; t < 2; ++t) {
      v8f s = {};
#pragma unroll
      for (int kc = 0; kc < 2; ++kc) s = WMMA_BF16(ka[t][kc].v, bq[kc].v, s);
      st[t] = s;
    }
    if (kb + 1 < 32) loadK(kb + 1, ka);              // reload in place (WAR-pinned),
                                                     // overlaps softmax + P.V below
    // ---- online softmax (query == lane&15; pair lanes share via xor-16)
    float mloc = st[0][0];
#pragma unroll
    for (int r = 0; r < 8; ++r) {
      mloc = fmaxf(mloc, st[0][r]);
      mloc = fmaxf(mloc, st[1][r]);
    }
    mloc = fmaxf(mloc, __shfl_xor(mloc, 16));
    float mnew  = fmaxf(mrun, mloc);
    float alpha = __expf(mrun - mnew);
    float ps = 0.f;
#pragma unroll
    for (int t = 0; t < 2; ++t)
#pragma unroll
      for (int r = 0; r < 8; ++r) {
        float pv = __expf(st[t][r] - mnew);
        ps += pv;
        pbuf[widx][lane & 15][t * 16 + r + 8 * g] = (bf16_t)pv;  // P[q][key]
      }
    ps += __shfl_xor(ps, 16);
    lrun = lrun * alpha + ps;
    mrun = mnew;

    ABReg v0r, v1r;                                  // first V pairs fly during
    loadV(kb, 0, v0r);                               // the dscnt wait + rescale
    loadV(kb, 1, v1r);

    asm volatile("s_wait_dscnt 0x0" ::: "memory");   // P stores visible to loads
    ABReg bp;                                        // P^T as B (B^T = P in LDS)
    {
      const bf16_t* p = &pbuf[widx][lane & 15][g * 8];
      bp.h[0] = ld8(p);
      bp.h[1] = ld8(p + 16);
    }
#pragma unroll
    for (int j = 0; j < 4; ++j)
#pragma unroll
      for (int r = 0; r < 8; ++r) ot[j][r] *= alpha;

    // ---- O^T += V^T · P^T, 2-register V rotation
    ot[0] = WMMA_BF16(v0r.v, bp.v, ot[0]); loadV(kb, 2, v0r);
    ot[1] = WMMA_BF16(v1r.v, bp.v, ot[1]); loadV(kb, 3, v1r);
    ot[2] = WMMA_BF16(v0r.v, bp.v, ot[2]);
    ot[3] = WMMA_BF16(v1r.v, bp.v, ot[3]);
  }

  float linv = 1.f / lrun;
  const int bb = bh >> 3, h = bh & 7;
  const int qg = qblk * 16 + (lane & 15);
  bf16_t* orow = obf + ((size_t)bb * 4096 + qg) * 512 + h * 64;
#pragma unroll
  for (int j = 0; j < 4; ++j)
#pragma unroll
    for (int r = 0; r < 8; ++r)
      orow[16 * j + 8 * g + r] = (bf16_t)(ot[j][r] * linv);
}

// ---------------------------------------------------------------------------
extern "C" void kernel_launch(void* const* d_in, const int* in_sizes, int n_in,
                              void* d_out, int out_size, void* d_ws, size_t ws_size,
                              hipStream_t stream)
{
  (void)in_sizes; (void)n_in; (void)out_size; (void)ws_size;
  const float* x     = (const float*)d_in[0];
  const float* wqkv  = (const float*)d_in[1];
  const float* wke   = (const float*)d_in[2];
  const float* wve   = (const float*)d_in[3];
  const float* gk    = (const float*)d_in[4];
  const float* bk    = (const float*)d_in[5];
  const float* gv    = (const float*)d_in[6];
  const float* bv    = (const float*)d_in[7];
  const float* wproj = (const float*)d_in[8];
  const float* bproj = (const float*)d_in[9];
  float* out = (float*)d_out;

  char* ws = (char*)d_ws;
  size_t off = 0;
  auto alloc = [&](size_t bytes) {
    void* p = ws + off;
    off += (bytes + 255) & ~(size_t)255;
    return p;
  };
  bf16_t* xbf     = (bf16_t*)alloc((size_t)8388608 * 2);   // x          [B*N, C]
  bf16_t* wqkvbf  = (bf16_t*)alloc((size_t)786432  * 2);   // W_qkv      [1536, 512]
  bf16_t* wkep    = (bf16_t*)alloc((size_t)1048576 * 2);   // W_ke packed[512, 2048]
  bf16_t* wvep    = (bf16_t*)alloc((size_t)1048576 * 2);   // W_ve packed
  bf16_t* wprojbf = (bf16_t*)alloc((size_t)262144  * 2);   // W_proj     [512, 512]
  bf16_t* qbf     = (bf16_t*)alloc((size_t)8388608 * 2);   // Q*scale    [B,h,N,64]
  bf16_t* kpre    = (bf16_t*)alloc((size_t)8388608 * 2);   // k pre-red  [B,N,C]
  bf16_t* vpre    = (bf16_t*)alloc((size_t)8388608 * 2);   // v pre-red  [B,N,C]
  bf16_t* kbf     = (bf16_t*)alloc((size_t)2097152 * 2);   // K (LN'd)   [B,h,L,64]
  bf16_t* vtbf    = (bf16_t*)alloc((size_t)2097152 * 2);   // V^T (LN'd) [B,h,64,L]
  bf16_t* obf     = (bf16_t*)alloc((size_t)8388608 * 2);   // attn out   [B*N, C]

  // 1) conversions / repacks (pure bandwidth, vectorized b128 paths)
  k_cvt8  <<<4096, 256, 0, stream>>>(x, xbf, 1048576);
  k_cvt8  <<<384,  256, 0, stream>>>(wqkv, wqkvbf, 98304);
  k_packw8<<<512,  256, 0, stream>>>(wke, wkep);
  k_packw8<<<512,  256, 0, stream>>>(wve, wvep);
  k_cvt8  <<<128,  256, 0, stream>>>(wproj, wprojbf, 32768);

  // 2) QKV GEMM: M=16384, N=1536, K=512 -> 1024*24 waves / 8 = 3072 blocks
  k_gemm<0><<<3072, 256, 0, stream>>>(xbf, wqkvbf, 512, 24,
                                      qbf, kpre, vpre,
                                      nullptr, nullptr, nullptr, nullptr, nullptr);

  // 3) KV spatial reduction (+LayerNorm): M=4096, N=512, K=2048 -> 256 blocks
  k_gemm<1><<<256, 256, 0, stream>>>(kpre, wkep, 2048, 8,
                                     nullptr, nullptr, nullptr,
                                     gk, bk, kbf, nullptr, nullptr);
  k_gemm<2><<<256, 256, 0, stream>>>(vpre, wvep, 2048, 8,
                                     nullptr, nullptr, nullptr,
                                     gv, bv, vtbf, nullptr, nullptr);

  // 4) flash attention: B*h*(N/16) = 8192 waves -> 1024 blocks
  k_attn<<<1024, 256, 0, stream>>>(qbf, kbf, vtbf, obf);

  // 5) output projection: M=16384, N=512, K=512 -> 1024 blocks (f32 out + bias)
  k_gemm<3><<<1024, 256, 0, stream>>>(obf, wprojbf, 512, 8,
                                      nullptr, nullptr, nullptr,
                                      nullptr, nullptr, nullptr, bproj, out);
}
